// Self_Attn_7851200217316
// MI455X (gfx1250) — matrix operand order repaired and checked by reading, not compiled, not run on hardware
//
#include <hip/hip_runtime.h>
#include <hip/hip_bf16.h>
#include <math.h>

typedef __attribute__((ext_vector_type(16))) __bf16 bf16x16;
typedef __attribute__((ext_vector_type(8)))  float  f32x8;

#define B_DIM 4
#define C_DIM 256
#define C8_DIM 32
#define N_DIM 4096
#define KBLK 64
#define NBLKS (N_DIM / KBLK)

// dynamic-LDS layout (ushort units): [kbuf0|kbuf1|vbuf0|vbuf1]
#define KROW 40                      // 64 keys x 32ch, padded row (80B, 16B-aligned)
#define VROW 72                      // 256 ch x 64 keys, padded row (144B, 16B-aligned)
#define KBUF_SZ (KBLK * KROW)        // 2560 ushorts
#define VBUF_SZ (C_DIM * VROW)       // 18432 ushorts
#define VBASE   (2 * KBUF_SZ)
#define DYN_LDS_BYTES ((2 * KBUF_SZ + 2 * VBUF_SZ) * 2)   // 83968 B

union ABFrag {
    bf16x16 v;
    int4 q[2];
};

__device__ __forceinline__ unsigned short f2bf(float f) {
    unsigned int u = __float_as_uint(f);
    unsigned int r = u + 0x7FFFu + ((u >> 16) & 1u);   // round-to-nearest-even
    return (unsigned short)(r >> 16);
}

// CDNA5 async copy: 16B global -> LDS, tracked by ASYNCcnt (no VGPR data path)
__device__ __forceinline__ void async_cp16(const unsigned short* gsrc,
                                           unsigned short* ldst) {
    unsigned lds = (unsigned)(uintptr_t)ldst;            // low 32 bits = LDS offset
    unsigned long long ga = (unsigned long long)(uintptr_t)gsrc;
    asm volatile("global_load_async_to_lds_b128 %0, %1, off"
                 :: "v"(lds), "v"(ga) : "memory");
}

__device__ __forceinline__ void wait_async0() {
    asm volatile("s_wait_asynccnt 0" ::: "memory");
}

// ---------------------------------------------------------------------------
// Kernel 1: 1x1-conv projections q,k,v  (fp32 in, bf16 out)
// q,k layout: [B][N][32] (row-major over channels)  v layout: [B][C][N]
// ---------------------------------------------------------------------------
__global__ __launch_bounds__(256) void proj_qkv(
    const float* __restrict__ x,                       // [B][C][N]
    const float* __restrict__ Wq, const float* __restrict__ bq,
    const float* __restrict__ Wk, const float* __restrict__ bk,
    const float* __restrict__ Wv, const float* __restrict__ bv,
    unsigned short* __restrict__ qb,
    unsigned short* __restrict__ kb,
    unsigned short* __restrict__ vb)
{
    const int n = blockIdx.x * 256 + threadIdx.x;      // 0..4095
    const int o = blockIdx.y;                          // 0..319
    const int b = blockIdx.z;

    const float* xcol = x + (size_t)b * C_DIM * N_DIM + n;
    const float* wrow;
    float acc;
    if (o < 32)       { wrow = Wq + o * C_DIM;        acc = bq[o]; }
    else if (o < 64)  { wrow = Wk + (o - 32) * C_DIM; acc = bk[o - 32]; }
    else              { wrow = Wv + (o - 64) * C_DIM; acc = bv[o - 64]; }

    #pragma unroll 8
    for (int c = 0; c < C_DIM; ++c)
        acc = fmaf(wrow[c], xcol[(size_t)c * N_DIM], acc);

    const unsigned short h = f2bf(acc);
    if (o < 32)
        qb[((size_t)(b * N_DIM + n)) * 32 + o] = h;
    else if (o < 64)
        kb[((size_t)(b * N_DIM + n)) * 32 + (o - 32)] = h;
    else
        vb[((size_t)(b * C_DIM + (o - 64))) * N_DIM + n] = h;
}

// ---------------------------------------------------------------------------
// Kernel 2: fused flash attention + gamma*out + x1
// 256 threads = 8 waves; wave -> (rowgroup rg = wave>>1, chalf = wave&1).
// Double-buffered K/V staging via GLOBAL_LOAD_ASYNC_TO_LDS (ASYNCcnt),
// so block nb+1's global fetch overlaps block nb's 160 WMMAs.
// ---------------------------------------------------------------------------
__global__ __launch_bounds__(256) void flash_attn(
    const unsigned short* __restrict__ qb,   // [B][N][32] bf16
    const unsigned short* __restrict__ kb,   // [B][N][32] bf16
    const unsigned short* __restrict__ vb,   // [B][C][N]  bf16
    const float* __restrict__ x1,            // [B][C][N]
    const float* __restrict__ gamma,
    float* __restrict__ out)                 // [B][C][N]
{
    extern __shared__ unsigned short dsm[];      // [kbuf0|kbuf1|vbuf0|vbuf1]
    __shared__ unsigned short plds[4][16][72];   // P tiles per rowgroup (9216 B)

    const int tid   = threadIdx.x;
    const int lane  = tid & 31;
    const int wave  = tid >> 5;
    const int rg    = wave >> 1;         // 0..3  (query row group)
    const int chalf = wave & 1;          // 0..1  (channel half)
    const int ln    = lane & 15;
    const int hi    = lane >> 4;

    const int b     = blockIdx.y;
    const int qrow0 = blockIdx.x * 64 + rg * 16;
    const float g   = gamma[0];

    // per-thread staging coordinates
    const int skey = tid >> 2, sch = (tid & 3) * 8;            // K: one 16B chunk
    const unsigned short* kgbase = kb + ((size_t)(b * N_DIM + skey)) * 32 + sch;
    const unsigned short* vgbase = vb + ((size_t)(b * C_DIM + tid)) * N_DIM;

    // Q A-fragment (16x32 bf16): lane row m=ln, K runs [hi*16,+16)
    ABFrag qa;
    {
        const unsigned short* qr = qb + ((size_t)(b * N_DIM + qrow0 + ln)) * 32;
        qa.q[0] = *(const int4*)(qr + hi * 16);
        qa.q[1] = *(const int4*)(qr + hi * 16 + 8);
    }

    f32x8 O[8];
    #pragma unroll
    for (int t = 0; t < 8; ++t) O[t] = 0.0f;
    float mrow[8], lrow[8];
    #pragma unroll
    for (int r = 0; r < 8; ++r) { mrow[r] = -1e30f; lrow[r] = 0.0f; }

    const f32x8 zacc = 0.0f;

    // ---- prologue: async-stage block 0 into buffer 0 ----
    async_cp16(kgbase, dsm + skey * KROW + sch);
    #pragma unroll
    for (int j = 0; j < 8; ++j)
        async_cp16(vgbase + j * 8, dsm + VBASE + tid * VROW + j * 8);

    for (int nb = 0; nb < NBLKS; ++nb) {
        const int cur = nb & 1;
        const int nxt = cur ^ 1;
        unsigned short* kcur = dsm + cur * KBUF_SZ;
        unsigned short* vcur = dsm + VBASE + cur * VBUF_SZ;

        wait_async0();       // my async copies for buf[cur] have landed in LDS
        __syncthreads();     // everyone's copies landed; buf[nxt] free to overwrite

        // ---- async-stage block nb+1 into buf[nxt] (overlaps compute) ----
        if (nb + 1 < NBLKS) {
            const int nk1 = (nb + 1) * KBLK;
            unsigned short* knxt = dsm + nxt * KBUF_SZ;
            unsigned short* vnxt = dsm + VBASE + nxt * VBUF_SZ;
            async_cp16(kgbase + (size_t)nk1 * 32, knxt + skey * KROW + sch);
            #pragma unroll
            for (int j = 0; j < 8; ++j)
                async_cp16(vgbase + nk1 + j * 8, vnxt + tid * VROW + j * 8);
        }

        // --- S = Q * K^T : 4 tiles of 16x16, one K=32 WMMA each ---
        f32x8 S[4];
        #pragma unroll
        for (int kt = 0; kt < 4; ++kt) {
            ABFrag kf;   // B-frag 32x16: lane col = key ln, K = ch [hi*16, hi*16+16)
            const unsigned short* kr = kcur + (kt * 16 + ln) * KROW + hi * 16;
            kf.q[0] = *(const int4*)kr;
            kf.q[1] = *(const int4*)(kr + 8);
            S[kt] = __builtin_amdgcn_wmma_f32_16x16x32_bf16(
                        false, qa.v, false, kf.v, (short)0, zacc, false, false);
        }

        // --- online softmax (row = r + 8*hi, cols across 16-lane half) ---
        float scale[8];
        #pragma unroll
        for (int r = 0; r < 8; ++r) {
            float mx = fmaxf(fmaxf(S[0][r], S[1][r]), fmaxf(S[2][r], S[3][r]));
            mx = fmaxf(mx, __shfl_xor(mx, 1));
            mx = fmaxf(mx, __shfl_xor(mx, 2));
            mx = fmaxf(mx, __shfl_xor(mx, 4));
            mx = fmaxf(mx, __shfl_xor(mx, 8));
            const float mnew = fmaxf(mrow[r], mx);
            scale[r] = __expf(mrow[r] - mnew);
            float rs = 0.0f;
            #pragma unroll
            for (int kt = 0; kt < 4; ++kt) {
                const float p = __expf(S[kt][r] - mnew);
                S[kt][r] = p;
                rs += p;
            }
            rs += __shfl_xor(rs, 1);
            rs += __shfl_xor(rs, 2);
            rs += __shfl_xor(rs, 4);
            rs += __shfl_xor(rs, 8);
            lrow[r] = lrow[r] * scale[r] + rs;
            mrow[r] = mnew;
        }
        #pragma unroll
        for (int t = 0; t < 8; ++t)
            #pragma unroll
            for (int r = 0; r < 8; ++r)
                O[t][r] = O[t][r] * scale[r];

        // --- P -> LDS in C/D layout (one wave per rg writes) ---
        if (chalf == 0) {
            #pragma unroll
            for (int kt = 0; kt < 4; ++kt)
                #pragma unroll
                for (int r = 0; r < 8; ++r)
                    plds[rg][r + 8 * hi][kt * 16 + ln] = f2bf(S[kt][r]);
        }
        __syncthreads();

        // --- O += P * V^T : 2 K-chunks x 8 channel tiles ---
        #pragma unroll
        for (int kc = 0; kc < 2; ++kc) {
            ABFrag pf;   // A-frag 16x32 over keys
            const unsigned short* pr = &plds[rg][ln][kc * 32 + hi * 16];
            pf.q[0] = *(const int4*)pr;
            pf.q[1] = *(const int4*)(pr + 8);
            #pragma unroll
            for (int ct = 0; ct < 8; ++ct) {
                ABFrag vf;   // B-frag 32x16: lane col = channel, K = keys [hi*16,+16)
                const unsigned short* vr =
                    vcur + (chalf * 128 + ct * 16 + ln) * VROW + kc * 32 + hi * 16;
                vf.q[0] = *(const int4*)vr;
                vf.q[1] = *(const int4*)(vr + 8);
                O[ct] = __builtin_amdgcn_wmma_f32_16x16x32_bf16(
                            false, pf.v, false, vf.v, (short)0, O[ct], false, false);
            }
        }
    }

    // --- epilogue: out = gamma * (O / l) + x1 ---
    float invl[8];
    #pragma unroll
    for (int r = 0; r < 8; ++r) invl[r] = 1.0f / lrow[r];

    #pragma unroll
    for (int ct = 0; ct < 8; ++ct) {
        const int c = chalf * 128 + ct * 16 + ln;
        const size_t base = ((size_t)(b * C_DIM + c)) * N_DIM + qrow0 + 8 * hi;
        #pragma unroll
        for (int r = 0; r < 8; ++r)
            out[base + r] = g * O[ct][r] * invl[r] + x1[base + r];
    }
}

// ---------------------------------------------------------------------------
extern "C" void kernel_launch(void* const* d_in, const int* in_sizes, int n_in,
                              void* d_out, int out_size, void* d_ws, size_t ws_size,
                              hipStream_t stream) {
    const float* x1    = (const float*)d_in[0];
    // d_in[1] = x2 : unused by the reference computation
    const float* Wq    = (const float*)d_in[2];
    const float* bq    = (const float*)d_in[3];
    const float* Wk    = (const float*)d_in[4];
    const float* bk    = (const float*)d_in[5];
    const float* Wv    = (const float*)d_in[6];
    const float* bv    = (const float*)d_in[7];
    const float* gamma = (const float*)d_in[8];
    float* out = (float*)d_out;

    unsigned short* qbw = (unsigned short*)d_ws;                  // 1 MB
    unsigned short* kbw = qbw + (size_t)B_DIM * N_DIM * C8_DIM;   // 1 MB
    unsigned short* vbw = kbw + (size_t)B_DIM * N_DIM * C8_DIM;   // 8 MB

    proj_qkv<<<dim3(N_DIM / 256, C8_DIM + C8_DIM + C_DIM, B_DIM), 256, 0, stream>>>(
        x1, Wq, bq, Wk, bk, Wv, bv, qbw, kbw, vbw);

    flash_attn<<<dim3(N_DIM / 64, B_DIM), 256, DYN_LDS_BYTES, stream>>>(
        qbw, kbw, vbw, x1, gamma, out);
}
